// DeformableTransformer_68101001445926
// MI455X (gfx1250) — matrix operand order repaired and checked by reading, compiled, not run on hardware
//
#include <hip/hip_runtime.h>

// ---------------------------------------------------------------------------
// Types for CDNA5 WMMA (gfx1250, wave32)
// ---------------------------------------------------------------------------
typedef __attribute__((ext_vector_type(16))) __bf16       v16bf;
typedef __attribute__((ext_vector_type(8)))  float        v8f;
typedef __attribute__((ext_vector_type(4)))  unsigned int u32x4;
typedef __attribute__((ext_vector_type(4)))  int          i32x4;

#define BATCH   2
#define LEN     5440
#define DM      256
#define DFFN    1024
#define NHEADS  8
#define HD      32
#define MROWS   (BATCH * LEN)   // 10880 = 85 * 128

// Async global->LDS path (CDNA5 GLOBAL_LOAD_ASYNC_TO_LDS_B128, ASYNCcnt)
#if defined(__gfx1250__) && __has_builtin(__builtin_amdgcn_global_load_async_to_lds_b128)
#define HAVE_ASYNC_LDS 1
#else
#define HAVE_ASYNC_LDS 0
#endif

#define AS1 __attribute__((address_space(1)))
#define AS3 __attribute__((address_space(3)))

// builtin expects: (i32x4 addrspace(1)*, i32x4 addrspace(3)*, imm offset, imm cpol)
__device__ __forceinline__ AS1 i32x4* to_global_v4(const void* p) {
  return (AS1 i32x4*)(unsigned long long)(size_t)p;
}
// Generic LDS pointer low 32 bits are the LDS byte offset (ISA 10.2 aperture map)
__device__ __forceinline__ AS3 i32x4* to_lds_v4(const void* p) {
  return (AS3 i32x4*)(unsigned int)(size_t)p;
}

__device__ __forceinline__ void wait_async0() {
#if defined(__gfx1250__)
#if __has_builtin(__builtin_amdgcn_s_wait_asynccnt)
  __builtin_amdgcn_s_wait_asynccnt(0);
#else
  asm volatile("s_wait_asynccnt 0" ::: "memory");
#endif
#endif
}

__device__ __forceinline__ unsigned short f2bf(float f) {
  unsigned int u = __float_as_uint(f);
  u += 0x7FFFu + ((u >> 16) & 1u);          // round-to-nearest-even
  return (unsigned short)(u >> 16);
}

union Frag { u32x4 u[2]; v16bf bf; };

// ---------------------------------------------------------------------------
// Weight prep: Wt[n*K + k] = bf16(W[k*N + n])   (transpose + convert)
// ---------------------------------------------------------------------------
__global__ void wprep_kernel(const float* __restrict__ W,
                             unsigned short* __restrict__ Wt, int K, int N) {
  int i = blockIdx.x * 256 + threadIdx.x;
  if (i >= K * N) return;
  int k = i / N, n = i - k * N;
  Wt[n * K + k] = f2bf(W[i]);
}

// ---------------------------------------------------------------------------
// Pack: Xbf = bf16(X), Qbf = bf16(X + pos)
// ---------------------------------------------------------------------------
__global__ void pack_kernel(const float* __restrict__ X,
                            const float* __restrict__ P,
                            unsigned short* __restrict__ Xbf,
                            unsigned short* __restrict__ Qbf, int n) {
  int i = blockIdx.x * 256 + threadIdx.x;
  if (i >= n) return;
  float x = X[i];
  Xbf[i] = f2bf(x);
  Qbf[i] = f2bf(x + P[i]);
}

// ---------------------------------------------------------------------------
// WMMA GEMM: C[M][N] = act(A_bf16[M][K] @ Bt_bf16[N][K]^T + bias)
// Block: 256 threads (8 waves, 4x2), tile 128x128, K-step 32.
// Wave tile 32x64 -> 2x4 v_wmma_f32_16x16x32_bf16 per K-step.
// Tiles staged via GLOBAL_LOAD_ASYNC_TO_LDS_B128 when available.
// ---------------------------------------------------------------------------
#define BM 128
#define BN 128
#define BK 32

__global__ __launch_bounds__(256)
void gemm_bf16_kernel(const unsigned short* __restrict__ A,
                      const unsigned short* __restrict__ Bt,
                      const float* __restrict__ bias,
                      float* __restrict__ Cf,
                      unsigned short* __restrict__ Cbf,
                      int M, int N, int K, int relu) {
  __shared__ __align__(16) unsigned short lsA[BM * BK];
  __shared__ __align__(16) unsigned short lsB[BN * BK];

  const int tid  = threadIdx.x;
  const int lane = tid & 31;
  const int wave = tid >> 5;
  const int wm   = wave >> 1;                 // 0..3  (rows)
  const int wn   = wave & 1;                  // 0..1  (cols)
  const int bm0  = blockIdx.y * BM;
  const int bn0  = blockIdx.x * BN;
  const int half = lane >> 4;
  const int l15  = lane & 15;

  v8f acc[2][4];
  v8f zero = {0.f, 0.f, 0.f, 0.f, 0.f, 0.f, 0.f, 0.f};
#pragma unroll
  for (int i = 0; i < 2; ++i)
#pragma unroll
    for (int j = 0; j < 4; ++j) acc[i][j] = zero;

  // staging chunk map: 512 16B-chunks per tile, 2 per thread
  const int r0 = tid >> 2, c0 = (tid & 3) * 8;          // chunk tid
  const int r1 = (tid + 256) >> 2, c1 = c0;             // chunk tid+256

  for (int k0 = 0; k0 < K; k0 += BK) {
#if HAVE_ASYNC_LDS
    __builtin_amdgcn_global_load_async_to_lds_b128(
        to_global_v4(&A[(size_t)(bm0 + r0) * K + k0 + c0]), to_lds_v4(&lsA[r0 * BK + c0]), 0, 0);
    __builtin_amdgcn_global_load_async_to_lds_b128(
        to_global_v4(&A[(size_t)(bm0 + r1) * K + k0 + c1]), to_lds_v4(&lsA[r1 * BK + c1]), 0, 0);
    __builtin_amdgcn_global_load_async_to_lds_b128(
        to_global_v4(&Bt[(size_t)(bn0 + r0) * K + k0 + c0]), to_lds_v4(&lsB[r0 * BK + c0]), 0, 0);
    __builtin_amdgcn_global_load_async_to_lds_b128(
        to_global_v4(&Bt[(size_t)(bn0 + r1) * K + k0 + c1]), to_lds_v4(&lsB[r1 * BK + c1]), 0, 0);
    wait_async0();
#else
    *(u32x4*)&lsA[r0 * BK + c0] = *(const u32x4*)&A[(size_t)(bm0 + r0) * K + k0 + c0];
    *(u32x4*)&lsA[r1 * BK + c1] = *(const u32x4*)&A[(size_t)(bm0 + r1) * K + k0 + c1];
    *(u32x4*)&lsB[r0 * BK + c0] = *(const u32x4*)&Bt[(size_t)(bn0 + r0) * K + k0 + c0];
    *(u32x4*)&lsB[r1 * BK + c1] = *(const u32x4*)&Bt[(size_t)(bn0 + r1) * K + k0 + c1];
#endif
    if (k0 + BK < K) {                        // prefetch next K tiles into L2
      __builtin_prefetch(&A[(size_t)(bm0 + r0) * K + k0 + BK + c0], 0, 1);
      __builtin_prefetch(&Bt[(size_t)(bn0 + r0) * K + k0 + BK + c0], 0, 1);
    }
    __syncthreads();

    // --- build fragments (ISA 7.12.2 16-bit layouts) --------------------
    Frag fa[2], fb[4];
#pragma unroll
    for (int i = 0; i < 2; ++i) {
      int row  = wm * 32 + i * 16 + l15;
      int base = row * BK + half * 8;          // K {0..7 | 8..15}
      fa[i].u[0] = *(const u32x4*)&lsA[base];
      fa[i].u[1] = *(const u32x4*)&lsA[base + 16];   // K {16..23 | 24..31}
    }
#pragma unroll
    for (int j = 0; j < 4; ++j) {
      int col  = wn * 64 + j * 16 + l15;
      int base = col * BK + half * 8;          // K {0..7 | 8..15}
      fb[j].u[0] = *(const u32x4*)&lsB[base];
      fb[j].u[1] = *(const u32x4*)&lsB[base + 16];   // K {16..23 | 24..31}
    }

#pragma unroll
    for (int i = 0; i < 2; ++i)
#pragma unroll
      for (int j = 0; j < 4; ++j)
        acc[i][j] = __builtin_amdgcn_wmma_f32_16x16x32_bf16(
            false, fa[i].bf, false, fb[j].bf, (short)0, acc[i][j],
            false, false);
    __syncthreads();
  }

  // --- epilogue: bias (+ReLU), write f32 and/or bf16 ----------------------
#pragma unroll
  for (int i = 0; i < 2; ++i)
#pragma unroll
    for (int j = 0; j < 4; ++j) {
      int col = bn0 + wn * 64 + j * 16 + l15;
      float bc = bias ? bias[col] : 0.f;
#pragma unroll
      for (int v = 0; v < 8; ++v) {
        int   row = bm0 + wm * 32 + i * 16 + v + 8 * half;
        float val = acc[i][j][v] + bc;
        if (relu) val = fmaxf(val, 0.f);
        size_t o = (size_t)row * N + col;
        if (Cf)  Cf[o]  = val;
        if (Cbf) Cbf[o] = f2bf(val);
      }
    }
}

// ---------------------------------------------------------------------------
// Softmax over 16 (levels*points) per (token, head)
// ---------------------------------------------------------------------------
__global__ void softmax16_kernel(float* __restrict__ AW, int MH) {
  int t = blockIdx.x * 256 + threadIdx.x;
  if (t >= MH) return;
  float* p = AW + (size_t)t * 16;
  float mx = -3.4e38f;
#pragma unroll
  for (int i = 0; i < 16; ++i) mx = fmaxf(mx, p[i]);
  float e[16], s = 0.f;
#pragma unroll
  for (int i = 0; i < 16; ++i) { e[i] = __expf(p[i] - mx); s += e[i]; }
  float inv = 1.f / s;
#pragma unroll
  for (int i = 0; i < 16; ++i) p[i] = e[i] * inv;
}

// ---------------------------------------------------------------------------
// Deformable sampling: one wave per (token, head), lane = head-dim channel.
// Value gathers are lane-contiguous (128B coalesced). Emits bf16 directly.
// ---------------------------------------------------------------------------
__global__ __launch_bounds__(256)
void deform_sample_kernel(const float* __restrict__ V,
                          const float* __restrict__ OFF,
                          const float* __restrict__ AW,
                          const float* __restrict__ vr,   // [B][4][2]
                          unsigned short* __restrict__ ATTbf) {
  const int Hs[4]  = {64, 32, 16, 8};
  const int Ws[4]  = {64, 32, 16, 8};
  const int Ss[4]  = {0, 4096, 5120, 5376};
  const int Lg2[4] = {6, 5, 4, 3};

  int wid  = blockIdx.x * 8 + (threadIdx.x >> 5);
  int lane = threadIdx.x & 31;
  if (wid >= MROWS * NHEADS) return;
  int m = wid >> 3, h = wid & 7;
  int b = m / LEN, q = m - b * LEN;

  // reference point of query q (its own level)
  int lq = (q < 4096) ? 0 : (q < 5120) ? 1 : (q < 5376) ? 2 : 3;
  int ql = q - Ss[lq];
  int colq = ql & (Ws[lq] - 1), rowq = ql >> Lg2[lq];
  float rx = (colq + 0.5f) / (vr[b * 8 + lq * 2 + 0] * (float)Ws[lq]);
  float ry = (rowq + 0.5f) / (vr[b * 8 + lq * 2 + 1] * (float)Hs[lq]);

  const float* offp = OFF + (size_t)m * DM + h * 32;  // [lvl][pt][xy]
  const float* awp  = AW + (size_t)m * (NHEADS * 16) + h * 16;

  float acc = 0.f;
#pragma unroll
  for (int lvl = 0; lvl < 4; ++lvl) {
    const int Hh = Hs[lvl], Ww = Ws[lvl];
    float fx = rx * vr[b * 8 + lvl * 2 + 0] * (float)Ww;
    float fy = ry * vr[b * 8 + lvl * 2 + 1] * (float)Hh;
    const float* vb = V + ((size_t)(b * LEN + Ss[lvl])) * DM + h * HD + lane;

    auto corner = [&](int cx, int cy, float w) -> float {
      bool valid = (cx >= 0) & (cx < Ww) & (cy >= 0) & (cy < Hh);
      int ccx = cx < 0 ? 0 : (cx > Ww - 1 ? Ww - 1 : cx);
      int ccy = cy < 0 ? 0 : (cy > Hh - 1 ? Hh - 1 : cy);
      float val = vb[(size_t)(ccy * Ww + ccx) * DM];
      return valid ? val * w : 0.f;
    };

#pragma unroll
    for (int p = 0; p < 4; ++p) {
      float x = fx + offp[lvl * 8 + p * 2 + 0] - 0.5f;
      float y = fy + offp[lvl * 8 + p * 2 + 1] - 0.5f;
      float x0f = floorf(x), y0f = floorf(y);
      float wx = x - x0f,    wy = y - y0f;
      int ix = (int)x0f, iy = (int)y0f;
      float a = awp[lvl * 4 + p];
      float s = corner(ix,     iy,     (1.f - wx) * (1.f - wy))
              + corner(ix + 1, iy,     wx * (1.f - wy))
              + corner(ix,     iy + 1, (1.f - wx) * wy)
              + corner(ix + 1, iy + 1, wx * wy);
      acc += s * a;
    }
  }
  ATTbf[(size_t)m * DM + h * HD + lane] = f2bf(acc);
}

// ---------------------------------------------------------------------------
// Residual + LayerNorm: one wave32 per 256-wide row, 8 rows / block.
// O = LN(A + B) * g + be ; optional bf16 copy.
// ---------------------------------------------------------------------------
__global__ __launch_bounds__(256)
void residual_ln_kernel(const float* __restrict__ A,
                        const float* __restrict__ B,
                        const float* __restrict__ g,
                        const float* __restrict__ be,
                        float* __restrict__ O,
                        unsigned short* __restrict__ Obf, int M) {
  int row  = blockIdx.x * 8 + (threadIdx.x >> 5);
  int lane = threadIdx.x & 31;
  if (row >= M) return;
  const float* a = A + (size_t)row * DM;
  const float* b = B + (size_t)row * DM;

  float x[8], s = 0.f, sq = 0.f;
#pragma unroll
  for (int i = 0; i < 8; ++i) {
    int c = i * 32 + lane;
    x[i] = a[c] + b[c];
    s  += x[i];
    sq += x[i] * x[i];
  }
#pragma unroll
  for (int off = 16; off > 0; off >>= 1) {
    s  += __shfl_xor(s,  off, 32);
    sq += __shfl_xor(sq, off, 32);
  }
  float mean = s * (1.f / DM);
  float var  = sq * (1.f / DM) - mean * mean;
  float inv  = rsqrtf(var + 1e-5f);
#pragma unroll
  for (int i = 0; i < 8; ++i) {
    int c = i * 32 + lane;
    float o = (x[i] - mean) * inv * g[c] + be[c];
    O[(size_t)row * DM + c] = o;
    if (Obf) Obf[(size_t)row * DM + c] = f2bf(o);
  }
}

// ---------------------------------------------------------------------------
// Host orchestration
// ---------------------------------------------------------------------------
extern "C" void kernel_launch(void* const* d_in, const int* in_sizes, int n_in,
                              void* d_out, int out_size, void* d_ws, size_t ws_size,
                              hipStream_t stream) {
  const float* src  = (const float*)d_in[0];
  const float* pos  = (const float*)d_in[1];
  const float* vr   = (const float*)d_in[2];
  const float* Wv   = (const float*)d_in[3];
  const float* bv   = (const float*)d_in[4];
  const float* Woff = (const float*)d_in[5];
  const float* boff = (const float*)d_in[6];
  const float* Wa   = (const float*)d_in[7];
  const float* ba   = (const float*)d_in[8];
  const float* Wo   = (const float*)d_in[9];
  const float* bo   = (const float*)d_in[10];
  const float* g1   = (const float*)d_in[11];
  const float* be1  = (const float*)d_in[12];
  const float* Wl1  = (const float*)d_in[13];
  const float* bl1  = (const float*)d_in[14];
  const float* Wl2  = (const float*)d_in[15];
  const float* bl2  = (const float*)d_in[16];
  const float* g2   = (const float*)d_in[17];
  const float* be2  = (const float*)d_in[18];

  const int M = MROWS;
  size_t cur = 0;
  auto carve = [&](size_t bytes) -> char* {
    char* p = (char*)d_ws + cur;
    cur += (bytes + 255) & ~(size_t)255;
    return p;
  };

  // bf16 transposed weights, both layers
  unsigned short* wtV[2];  unsigned short* wtOf[2]; unsigned short* wtA[2];
  unsigned short* wtO[2];  unsigned short* wtL1[2]; unsigned short* wtL2[2];
  for (int l = 0; l < 2; ++l) {
    wtV[l]  = (unsigned short*)carve((size_t)DM * DM * 2);
    wtOf[l] = (unsigned short*)carve((size_t)DM * DM * 2);
    wtA[l]  = (unsigned short*)carve((size_t)DM * 128 * 2);
    wtO[l]  = (unsigned short*)carve((size_t)DM * DM * 2);
    wtL1[l] = (unsigned short*)carve((size_t)DM * DFFN * 2);
    wtL2[l] = (unsigned short*)carve((size_t)DFFN * DM * 2);
  }
  unsigned short* Xbf   = (unsigned short*)carve((size_t)M * DM * 2);
  unsigned short* Qbf   = (unsigned short*)carve((size_t)M * DM * 2);   // reused as X2bf
  float*          Vbuf  = (float*)carve((size_t)M * DM * 4);            // reused as X2
  float*          OFFb  = (float*)carve((size_t)M * DM * 4);            // reused as ATT
  float*          AWb   = (float*)carve((size_t)M * 128 * 4);
  unsigned short* ATTbf = (unsigned short*)carve((size_t)M * DM * 2);
  unsigned short* Hbf   = (unsigned short*)carve((size_t)M * DFFN * 2);
  float*          Fbuf  = (float*)carve((size_t)M * DM * 4);
  float*          Sout  = (float*)carve((size_t)M * DM * 4);
  (void)ws_size; (void)n_in; (void)in_sizes; (void)out_size;

  // --- one-time weight transpose/convert ---------------------------------
  for (int l = 0; l < 2; ++l) {
    wprep_kernel<<<(DM * DM + 255) / 256, 256, 0, stream>>>(Wv + (size_t)l * DM * DM, wtV[l], DM, DM);
    wprep_kernel<<<(DM * DM + 255) / 256, 256, 0, stream>>>(Woff + (size_t)l * DM * DM, wtOf[l], DM, DM);
    wprep_kernel<<<(DM * 128 + 255) / 256, 256, 0, stream>>>(Wa + (size_t)l * DM * 128, wtA[l], DM, 128);
    wprep_kernel<<<(DM * DM + 255) / 256, 256, 0, stream>>>(Wo + (size_t)l * DM * DM, wtO[l], DM, DM);
    wprep_kernel<<<(DM * DFFN + 255) / 256, 256, 0, stream>>>(Wl1 + (size_t)l * DM * DFFN, wtL1[l], DM, DFFN);
    wprep_kernel<<<(DFFN * DM + 255) / 256, 256, 0, stream>>>(Wl2 + (size_t)l * DFFN * DM, wtL2[l], DFFN, DM);
  }

  const int nElem = M * DM;
  for (int l = 0; l < 2; ++l) {
    const float* Xin  = (l == 0) ? src : Sout;
    float*       Xmid = Vbuf;                       // reuse V region after sampling
    float*       ATT  = OFFb;                       // reuse OFF region after sampling
    unsigned short* X2bf = Qbf;                     // reuse Qbf after off/aw GEMMs
    float*       Xout = (l == 1) ? (float*)d_out : Sout;

    pack_kernel<<<(nElem + 255) / 256, 256, 0, stream>>>(Xin, pos, Xbf, Qbf, nElem);

    // value = X @ Wv + bv
    gemm_bf16_kernel<<<dim3(DM / BN, M / BM), 256, 0, stream>>>(
        Xbf, wtV[l], bv + l * DM, Vbuf, nullptr, M, DM, DM, 0);
    // offsets = Q @ Woff + boff
    gemm_bf16_kernel<<<dim3(DM / BN, M / BM), 256, 0, stream>>>(
        Qbf, wtOf[l], boff + l * DM, OFFb, nullptr, M, DM, DM, 0);
    // attn logits = Q @ Wa + ba
    gemm_bf16_kernel<<<dim3(128 / BN, M / BM), 256, 0, stream>>>(
        Qbf, wtA[l], ba + l * 128, AWb, nullptr, M, 128, DM, 0);

    softmax16_kernel<<<(M * NHEADS + 255) / 256, 256, 0, stream>>>(AWb, M * NHEADS);

    deform_sample_kernel<<<M, 256, 0, stream>>>(Vbuf, OFFb, AWb, vr, ATTbf);

    // attn_out = sampled @ Wo + bo
    gemm_bf16_kernel<<<dim3(DM / BN, M / BM), 256, 0, stream>>>(
        ATTbf, wtO[l], bo + l * DM, ATT, nullptr, M, DM, DM, 0);

    // X2 = LN(Xin + attn_out)
    residual_ln_kernel<<<M / 8, 256, 0, stream>>>(Xin, ATT, g1 + l * DM, be1 + l * DM,
                                                  Xmid, X2bf, M);
    // H = relu(X2 @ Wl1 + bl1)  -> bf16 only
    gemm_bf16_kernel<<<dim3(DFFN / BN, M / BM), 256, 0, stream>>>(
        X2bf, wtL1[l], bl1 + l * DFFN, nullptr, Hbf, M, DFFN, DM, 1);
    // F = H @ Wl2 + bl2
    gemm_bf16_kernel<<<dim3(DM / BN, M / BM), 256, 0, stream>>>(
        Hbf, wtL2[l], bl2 + l * DM, Fbuf, nullptr, M, DM, DFFN, 0);
    // Xout = LN(X2 + F)
    residual_ln_kernel<<<M / 8, 256, 0, stream>>>(Xmid, Fbuf, g2 + l * DM, be2 + l * DM,
                                                  Xout, nullptr, M);
  }
}
